// WindowAttention_59940563583748
// MI455X (gfx1250) — matrix operand order repaired and checked by reading, not compiled, not run on hardware
//
#include <hip/hip_runtime.h>

typedef __attribute__((ext_vector_type(16))) __bf16 v16bf;
typedef __attribute__((ext_vector_type(8)))  __bf16 v8bf;
typedef __attribute__((ext_vector_type(8)))  float  v8f;

#define WMMA_BF16(a, b, c) \
  __builtin_amdgcn_wmma_f32_16x16x32_bf16(false, (a), false, (b), (short)0, (c), false, false)

// ---- fragment loaders (ISA 7.12.2 layouts) -------------------------------
// A (16-bit, 16x32): lane holds row M=lane%16; elements 0..7 = K (hi*8)..(hi*8+7),
// elements 8..15 = K (16+hi*8)..(23+hi*8).  p must point at row + k0 + hi*8.
__device__ __forceinline__ v16bf ldA(const __bf16* p) {
  union { v16bf v; v8bf h[2]; } u;
  u.h[0] = *(const v8bf*)(p);
  u.h[1] = *(const v8bf*)(p + 16);
  return u.v;
}
// B (32x16) from an [N][K] (transposed) matrix: lane holds col N=lane%16;
// elements 0..7 = K (k0+hi*8)..(k0+hi*8+7), elements 8..15 = K (k0+16+hi*8)..(k0+23+hi*8).  p = row(N) + k0 + hi*8.
__device__ __forceinline__ v16bf ldB(const __bf16* p) {
  union { v16bf v; v8bf h[2]; } u;
  u.h[0] = *(const v8bf*)(p);
  u.h[1] = *(const v8bf*)(p + 16);
  return u.v;
}

// ---- conversion kernels ---------------------------------------------------
__global__ void k_f32_to_bf16(const float* __restrict__ in, __bf16* __restrict__ out, long n) {
  long i = (long)blockIdx.x * blockDim.x + threadIdx.x;
  long stride = (long)gridDim.x * blockDim.x;
  for (; i < n; i += stride) out[i] = (__bf16)in[i];
}

// out[c][r] = bf16(in[r][c]);  in is rows x cols, out is cols x rows.
__global__ void k_transpose_bf16(const float* __restrict__ in, __bf16* __restrict__ out,
                                 int rows, int cols) {
  long n = (long)rows * cols;
  long i = (long)blockIdx.x * blockDim.x + threadIdx.x;
  long stride = (long)gridDim.x * blockDim.x;
  for (; i < n; i += stride) {
    int c = (int)(i / rows);
    int r = (int)(i - (long)c * rows);
    out[i] = (__bf16)in[(long)r * cols + c];
  }
}

// ---- kernel 1: QKV GEMM per window ---------------------------------------
// x window rows (64x384 bf16) @ WtQKV (1152x384, N-major) -> Q,K scaled/copied,
// V written transposed: Vt[win][h][d][token].
__global__ __launch_bounds__(256) void k_qkv(const __bf16* __restrict__ xbf,
                                             const __bf16* __restrict__ WtQKV,
                                             __bf16* __restrict__ Qb,
                                             __bf16* __restrict__ Kb,
                                             __bf16* __restrict__ Vt) {
  const int win  = blockIdx.x;
  const int tid  = threadIdx.x;
  const int wave = tid >> 5, lane = tid & 31;
  const int hi   = lane >> 4, ln = lane & 15;
  const int rt   = wave & 3;      // row tile (16 tokens)
  const int half = wave >> 2;     // column half: 36 col tiles each

  // window-partition gather for this lane's A row
  const int tok = rt * 16 + ln;
  const int b = win >> 6, wI = win & 63;
  const int gy = ((wI >> 3) << 3) + (tok >> 3);
  const int gx = ((wI & 7) << 3) + (tok & 7);
  const __bf16* rp = xbf + (((size_t)b * 64 + gy) * 64 + gx) * 384;

  v16bf afr[12];
#pragma unroll
  for (int kk = 0; kk < 12; ++kk) afr[kk] = ldA(rp + kk * 32 + hi * 8);

  const float qscale = 0.17677669529663687f;  // (C/nh)^-0.5 = 32^-0.5
  for (int cti = 0; cti < 36; ++cti) {
    const int ct = half * 36 + cti;
    const int c = ct * 16 + ln;                 // output column 0..1151
    v8f acc = {};
#pragma unroll
    for (int kk = 0; kk < 12; ++kk) {
      v16bf bf = ldB(WtQKV + (size_t)c * 384 + kk * 32 + hi * 8);
      acc = WMMA_BF16(afr[kk], bf, acc);
    }
    const int s = c / 384;                      // 0=Q 1=K 2=V
    const int rem = c - s * 384;
    const int h = rem >> 5, d = rem & 31;
    const size_t hb = (size_t)win * 12 + h;
#pragma unroll
    for (int j = 0; j < 8; ++j) {
      const int t = rt * 16 + j + 8 * hi;
      const float v = acc[j];
      if (s == 0)      Qb[(hb * 64 + t) * 32 + d] = (__bf16)(v * qscale);
      else if (s == 1) Kb[(hb * 64 + t) * 32 + d] = (__bf16)v;
      else             Vt[(hb * 32 + d) * 64 + t] = (__bf16)v;
    }
  }
}

// ---- kernel 2: attention per (window, head) ------------------------------
__global__ __launch_bounds__(128) void k_attn(const __bf16* __restrict__ Qb,
                                              const __bf16* __restrict__ Kb,
                                              const __bf16* __restrict__ Vt,
                                              __bf16* __restrict__ Ob) {
  __shared__ __align__(16) float  S[64][68];   // padded to dodge bank conflicts
  __shared__ __align__(16) __bf16 P[64][72];   // 144B row stride, 16B aligned chunks

  const int bid = blockIdx.x;
  const int win = bid / 12, h = bid - win * 12;
  const int tid = threadIdx.x;
  const int wave = tid >> 5, lane = tid & 31;
  const int hi = lane >> 4, ln = lane & 15;
  const int m0 = wave * 16;                     // 16 query rows per wave
  const size_t hb = (size_t)win * 12 + h;
  const __bf16* Qh = Qb + hb * 64 * 32;
  const __bf16* Kh = Kb + hb * 64 * 32;
  const __bf16* Vh = Vt + hb * 32 * 64;

  // S = Q K^T  (A = Q row-major, B = K^T -> K row-major is already [n][k])
  v16bf aq = ldA(Qh + (size_t)(m0 + ln) * 32 + hi * 8);
#pragma unroll
  for (int ct = 0; ct < 4; ++ct) {
    v16bf bk = ldB(Kh + (size_t)(ct * 16 + ln) * 32 + hi * 8);
    v8f acc = {};
    acc = WMMA_BF16(aq, bk, acc);
#pragma unroll
    for (int j = 0; j < 8; ++j) S[m0 + j + 8 * hi][ct * 16 + ln] = acc[j];
  }
  __syncthreads();

  // row-wise softmax (one thread per query row), P in bf16
  if (tid < 64) {
    float m = -1e30f;
#pragma unroll
    for (int j = 0; j < 64; ++j) m = fmaxf(m, S[tid][j]);
    float e[64];
    float sum = 0.f;
#pragma unroll
    for (int j = 0; j < 64; ++j) { e[j] = __expf(S[tid][j] - m); sum += e[j]; }
    const float inv = 1.f / sum;
#pragma unroll
    for (int j = 0; j < 64; ++j) P[tid][j] = (__bf16)(e[j] * inv);
  }
  __syncthreads();

  // O = P V  (A = P from LDS, B = V via transposed Vt[d][token])
#pragma unroll
  for (int nt = 0; nt < 2; ++nt) {
    v8f acc = {};
#pragma unroll
    for (int kt = 0; kt < 2; ++kt) {
      v16bf ap = ldA(&P[m0 + ln][kt * 32 + hi * 8]);
      v16bf bv = ldB(Vh + (size_t)(nt * 16 + ln) * 64 + kt * 32 + hi * 8);
      acc = WMMA_BF16(ap, bv, acc);
    }
#pragma unroll
    for (int j = 0; j < 8; ++j) {
      const int t = m0 + j + 8 * hi;
      Ob[((size_t)win * 64 + t) * 384 + h * 32 + nt * 16 + ln] = (__bf16)acc[j];
    }
  }
}

// ---- kernel 3: proj GEMM + bias + window_reverse scatter -----------------
__global__ __launch_bounds__(256) void k_proj(const __bf16* __restrict__ Ob,
                                              const __bf16* __restrict__ WtP,
                                              const float* __restrict__ bias,
                                              float* __restrict__ out) {
  const int win = blockIdx.x;
  const int tid = threadIdx.x;
  const int wave = tid >> 5, lane = tid & 31;
  const int hi = lane >> 4, ln = lane & 15;
  const int rt = wave & 3, half = wave >> 2;

  const int tokA = rt * 16 + ln;
  const __bf16* rp = Ob + ((size_t)win * 64 + tokA) * 384;
  v16bf afr[12];
#pragma unroll
  for (int kk = 0; kk < 12; ++kk) afr[kk] = ldA(rp + kk * 32 + hi * 8);

  const int b = win >> 6, wI = win & 63;
  const int y0 = (wI >> 3) << 3, x0 = (wI & 7) << 3;

  for (int cti = 0; cti < 12; ++cti) {
    const int ct = half * 12 + cti;
    const int c = ct * 16 + ln;
    v8f acc = {};
#pragma unroll
    for (int kk = 0; kk < 12; ++kk) {
      v16bf bf = ldB(WtP + (size_t)c * 384 + kk * 32 + hi * 8);
      acc = WMMA_BF16(afr[kk], bf, acc);
    }
    const float bc = bias[c];
#pragma unroll
    for (int j = 0; j < 8; ++j) {
      const int t = rt * 16 + j + 8 * hi;
      const int gy = y0 + (t >> 3), gx = x0 + (t & 7);
      out[(((size_t)b * 64 + gy) * 64 + gx) * 384 + c] = acc[j] + bc;
    }
  }
}

// ---- launch ---------------------------------------------------------------
extern "C" void kernel_launch(void* const* d_in, const int* in_sizes, int n_in,
                              void* d_out, int out_size, void* d_ws, size_t ws_size,
                              hipStream_t stream) {
  (void)in_sizes; (void)n_in; (void)out_size; (void)ws_size;
  const float* x      = (const float*)d_in[0];
  const float* w_qkv  = (const float*)d_in[1];
  const float* w_proj = (const float*)d_in[2];
  const float* b_proj = (const float*)d_in[3];
  float* out = (float*)d_out;

  char* ws = (char*)d_ws;
  size_t off = 0;
  auto take = [&](size_t elems) -> __bf16* {
    __bf16* p = (__bf16*)(ws + off);
    off += ((elems * 2 + 255) & ~(size_t)255);
    return p;
  };

  const long NX = 32L * 64 * 64 * 384;          // 50.33M elements
  const long NH = 2048L * 12 * 64 * 32;         // per Q/K/V buffer
  __bf16* xbf   = take(NX);
  __bf16* WtQKV = take(1152L * 384);
  __bf16* WtP   = take(384L * 384);
  __bf16* Qb    = take(NH);
  __bf16* Kb    = take(NH);
  __bf16* Vt    = take(NH);
  __bf16* Ob    = take(2048L * 64 * 384);

  k_f32_to_bf16<<<2048, 256, 0, stream>>>(x, xbf, NX);
  k_transpose_bf16<<<1728, 256, 0, stream>>>(w_qkv, WtQKV, 384, 1152);
  k_transpose_bf16<<<576, 256, 0, stream>>>(w_proj, WtP, 384, 384);
  k_qkv<<<2048, 256, 0, stream>>>(xbf, WtQKV, Qb, Kb, Vt);
  k_attn<<<2048 * 12, 128, 0, stream>>>(Qb, Kb, Vt, Ob);
  k_proj<<<2048, 256, 0, stream>>>(Ob, WtP, b_proj, out);
}
